// DIGNN_RW_62423054680275
// MI455X (gfx1250) — matrix operand order repaired and checked by reading, not compiled, not run on hardware
//
#include <hip/hip_runtime.h>
#include <hip/hip_bf16.h>

// ---------------------------------------------------------------------------
// DIGNN-RW forward for MI455X (gfx1250, wave32, WMMA).
//   Dense GEMMs  -> v_wmma_f32_16x16x32_bf16, pre-converted bf16 A (b128
//                   loads) + weights pre-packed into per-lane B fragments
//                   (one 32B load per fragment). K-loop fully unrolled.
//   Sparse prop  -> edge-parallel scatter with f32 global atomics
//                   (z is 51.2MB -> L2-resident on the 192MB L2)
// ---------------------------------------------------------------------------

#define N_NODES   100000
#define N_EDGES   1600000
#define N_GRAPHS  512
#define C_IN      128
#define HDIM      128
#define C_OUTV    10
#define MU_V      1.0f
#define MAX_ITER  10
#define BN_EPS    1e-5f

typedef __attribute__((ext_vector_type(16))) __bf16 v16bf;
typedef __attribute__((ext_vector_type(8)))  __bf16 v8bf;
typedef __attribute__((ext_vector_type(8)))  float  v8f;

__device__ __forceinline__ __bf16 f2bf(float f) { return (__bf16)f; }

__device__ __forceinline__ void atomAddF(float* p, float v) {
  __hip_atomic_fetch_add(p, v, __ATOMIC_RELAXED, __HIP_MEMORY_SCOPE_AGENT);
}

// ---------------------------------------------------------------------------
// f32 -> bf16 streaming convert (8 elements / thread, b128 in, b128 out)
// n must be a multiple of 8 (true for all uses here).
// ---------------------------------------------------------------------------
__global__ void cvt_f32_bf16_kernel(const float* __restrict__ s,
                                    __bf16* __restrict__ d, long long n) {
  long long i = ((long long)blockIdx.x * blockDim.x + threadIdx.x) * 8;
  if (i >= n) return;
  const float4 a = *(const float4*)(s + i);
  const float4 b = *(const float4*)(s + i + 4);
  v8bf o;
  o[0] = f2bf(a.x); o[1] = f2bf(a.y); o[2] = f2bf(a.z); o[3] = f2bf(a.w);
  o[4] = f2bf(b.x); o[5] = f2bf(b.y); o[6] = f2bf(b.z); o[7] = f2bf(b.w);
  *(v8bf*)(d + i) = o;
}

// ---------------------------------------------------------------------------
// Pack W[K,M] (f32, row-major) into per-lane WMMA B fragments (bf16):
//   Wp[((colTile*(K/32) + kb)*32 + lane)*16 + j]
//     = bf16( W[(kb*32 + (lane&16 ? 8:0) + j + (j>=8 ? 8:0))*M + colTile*16 + (lane&15)] )
// One thread per (colTile, kb, lane); 16 elements each. Tiny (<= 64KB).
// ---------------------------------------------------------------------------
__global__ void pack_w_kernel(const float* __restrict__ W,
                              __bf16* __restrict__ Wp, int K, int M) {
  const int K32 = K >> 5;
  const int total = (M >> 4) * K32 * 32;
  int t = blockIdx.x * blockDim.x + threadIdx.x;
  if (t >= total) return;
  const int lane = t & 31;
  const int kb   = (t >> 5) % K32;
  const int ct   = (t >> 5) / K32;
  const int kBase = kb * 32 + ((lane & 16) ? 8 : 0);
  const int col   = ct * 16 + (lane & 15);
  __bf16* dst = Wp + (size_t)t * 16;
#pragma unroll
  for (int j = 0; j < 16; ++j)
    dst[j] = f2bf(W[(size_t)(kBase + j + (j >= 8 ? 8 : 0)) * M + col]);
}

// ---------------------------------------------------------------------------
// WMMA GEMM: out[rows,M] = epilogue(A[rows,K] @ W[K,M] + bias)
// One wave per 16x16 tile; grid = (rows/16, M/16). K, M compile-time.
// MODE: 1 = bias+relu, 2 = bias then BatchNorm(eval) affine.
// OBF:  1 = store bf16 (consumer is another GEMM), 0 = store f32.
// A is bf16 row-major; Wp is pre-packed B fragments (see pack_w_kernel).
// Fragment layouts per CDNA5 ISA 7.12.2.
// ---------------------------------------------------------------------------
template <int MODE, int OBF, int K, int M>
__global__ __launch_bounds__(32) void wmma_gemm_kernel(
    const __bf16* __restrict__ A, const __bf16* __restrict__ Wp,
    const float* __restrict__ bias, void* __restrict__ outp,
    const float* __restrict__ bn_g, const float* __restrict__ bn_b,
    const float* __restrict__ bn_m, const float* __restrict__ bn_v)
{
  constexpr int K32 = K / 32;
  const int lane    = threadIdx.x;           // 0..31
  const int rowTile = blockIdx.x;
  const int colTile = blockIdx.y;
  const int m15     = lane & 15;
  const int kA      = (lane & 16) ? 8 : 0;   // A K-offset per lane half

  const __bf16* __restrict__ ap = A + (size_t)(rowTile * 16 + m15) * K;
  const v16bf*  __restrict__ bp =
      (const v16bf*)Wp + (size_t)colTile * K32 * 32 + lane;

  v8f acc = {};
#pragma unroll
  for (int kb = 0; kb < K32; ++kb) {
    // A 16x32 fragment: 8 bf16 at K=kb*32+kA, 8 bf16 at K=kb*32+kA+16
    const v8bf lo = *(const v8bf*)(ap + kb * 32 + kA);
    const v8bf hi = *(const v8bf*)(ap + kb * 32 + kA + 16);
    const v16bf af = __builtin_shufflevector(
        lo, hi, 0, 1, 2, 3, 4, 5, 6, 7, 8, 9, 10, 11, 12, 13, 14, 15);
    const v16bf bf_ = bp[(size_t)kb * 32];   // one aligned 32B fragment load
    acc = __builtin_amdgcn_wmma_f32_16x16x32_bf16(
        /*neg_a=*/false, af, /*neg_b=*/false, bf_,
        /*c_mod=*/(short)0, acc, /*reuse_a=*/false, /*reuse_b=*/false);
  }

  const int col = colTile * 16 + m15;
  const float bv = bias[col];
  float scale = 1.f, shift = 0.f;
  if (MODE == 2) {
    scale = rsqrtf(bn_v[col] + BN_EPS) * bn_g[col];
    shift = bn_b[col] - bn_m[col] * scale;
  }
  const int rbase = rowTile * 16 + ((lane & 16) ? 8 : 0);
#pragma unroll
  for (int r = 0; r < 8; ++r) {              // C/D: row = rbase+r, col = col
    float v = acc[r] + bv;
    if (MODE == 1) v = fmaxf(v, 0.f);
    if (MODE == 2) v = v * scale + shift;
    if (OBF)
      ((__bf16*)outp)[(size_t)(rbase + r) * M + col] = f2bf(v);
    else
      ((float*)outp)[(size_t)(rbase + r) * M + col] = v;
  }
}

// ---------------------------------------------------------------------------
// Utility / sparse kernels
// ---------------------------------------------------------------------------
__global__ void zero_kernel(float* __restrict__ p, long long n) {
  long long i = (long long)blockIdx.x * blockDim.x + threadIdx.x;
  if (i < n) p[i] = 0.f;
}

__global__ void deg_kernel(const int* __restrict__ row,
                           const float* __restrict__ ew,
                           float* __restrict__ deg, int E) {
  int e = blockIdx.x * blockDim.x + threadIdx.x;
  if (e < E) atomAddF(&deg[row[e]], ew[e]);
}

__global__ void recip_kernel(float* __restrict__ deg, int n) {
  int i = blockIdx.x * blockDim.x + threadIdx.x;
  if (i < n) deg[i] = 1.0f / fmaxf(deg[i], 1e-12f);
}

// msg = ew[e] * z[col[e]]; agg[row[e]] += msg.  32 lanes per edge, 4 ch each.
__global__ void scatter_kernel(const int* __restrict__ row,
                               const int* __restrict__ col,
                               const float* __restrict__ ew,
                               const float* __restrict__ z,
                               float* __restrict__ agg, int E) {
  long long t = (long long)blockIdx.x * blockDim.x + threadIdx.x;
  int e = (int)(t >> 5);
  int c = ((int)t & 31) * 4;
  if (e >= E) return;
  const int r = row[e], cl = col[e];
  const float w = ew[e];
  const float4 zv = *(const float4*)(z + (size_t)cl * HDIM + c);
  float* dst = agg + (size_t)r * HDIM + c;
  atomAddF(dst + 0, w * zv.x);
  atomAddF(dst + 1, w * zv.y);
  atomAddF(dst + 2, w * zv.z);
  atomAddF(dst + 3, w * zv.w);
}

// in-place: z[i] = (z[i] * deg_inv[node] + MU*h[i]) / (1+MU)
__global__ void combine_kernel(float* __restrict__ z,
                               const float* __restrict__ deg_inv,
                               const float* __restrict__ h, long long n) {
  long long i = (long long)blockIdx.x * blockDim.x + threadIdx.x;
  if (i < n) {
    const float di = deg_inv[i >> 7];  // HDIM == 128
    z[i] = (z[i] * di + MU_V * h[i]) * (1.0f / (1.0f + MU_V));
  }
}

// global_add_pool: g[batch[n]] += z[n].  32 lanes per node, 4 ch each.
__global__ void pool_kernel(const float* __restrict__ z,
                            const int* __restrict__ batch,
                            float* __restrict__ g, int N) {
  long long t = (long long)blockIdx.x * blockDim.x + threadIdx.x;
  int n = (int)(t >> 5);
  int c = ((int)t & 31) * 4;
  if (n >= N) return;
  const int b = batch[n];
  const float4 zv = *(const float4*)(z + (size_t)n * HDIM + c);
  float* dst = g + (size_t)b * HDIM + c;
  atomAddF(dst + 0, zv.x);
  atomAddF(dst + 1, zv.y);
  atomAddF(dst + 2, zv.z);
  atomAddF(dst + 3, zv.w);
}

// out = log_softmax(g @ final_w + final_b), one thread per graph (G=512 tiny)
__global__ void final_kernel(const float* __restrict__ g,
                             const float* __restrict__ fw,
                             const float* __restrict__ fb,
                             float* __restrict__ out, int G) {
  int gi = blockIdx.x * blockDim.x + threadIdx.x;
  if (gi >= G) return;
  float o[C_OUTV];
#pragma unroll
  for (int j = 0; j < C_OUTV; ++j) o[j] = fb[j];
  const float* gr = g + (size_t)gi * HDIM;
  for (int k = 0; k < HDIM; ++k) {
    const float v = gr[k];
#pragma unroll
    for (int j = 0; j < C_OUTV; ++j) o[j] += v * fw[k * C_OUTV + j];
  }
  float m = o[0];
#pragma unroll
  for (int j = 1; j < C_OUTV; ++j) m = fmaxf(m, o[j]);
  float s = 0.f;
#pragma unroll
  for (int j = 0; j < C_OUTV; ++j) s += __expf(o[j] - m);
  const float ls = __logf(s);
#pragma unroll
  for (int j = 0; j < C_OUTV; ++j) out[gi * C_OUTV + j] = o[j] - m - ls;
}

// ---------------------------------------------------------------------------
// Host orchestration
// ---------------------------------------------------------------------------
static inline void launch_zero(float* p, long long n, hipStream_t s) {
  zero_kernel<<<(unsigned)((n + 255) / 256), 256, 0, s>>>(p, n);
}
static inline void launch_cvt(const float* s, __bf16* d, long long n,
                              hipStream_t st) {
  cvt_f32_bf16_kernel<<<(unsigned)((n / 8 + 255) / 256), 256, 0, st>>>(s, d, n);
}
static inline void launch_pack(const float* w, __bf16* wp, int K, int M,
                               hipStream_t st) {
  int total = (M >> 4) * (K >> 5) * 32;
  pack_w_kernel<<<(total + 255) / 256, 256, 0, st>>>(w, wp, K, M);
}

extern "C" void kernel_launch(void* const* d_in, const int* in_sizes, int n_in,
                              void* d_out, int out_size, void* d_ws, size_t ws_size,
                              hipStream_t stream) {
  // setup_inputs() dict order:
  const float* x     = (const float*)d_in[0];   // [N, 128]
  const int*   eidx  = (const int*)  d_in[1];   // [2, E]
  const float* ew    = (const float*)d_in[2];   // [E]
  const int*   batch = (const int*)  d_in[3];   // [N]
  const float* w1    = (const float*)d_in[4];   // [128, 64]
  const float* b1    = (const float*)d_in[5];
  const float* w2    = (const float*)d_in[6];   // [64, 128]
  const float* b2    = (const float*)d_in[7];
  const float* w3    = (const float*)d_in[8];   // [128, 128]
  const float* b3    = (const float*)d_in[9];
  const float* bng   = (const float*)d_in[10];
  const float* bnb   = (const float*)d_in[11];
  const float* bnm   = (const float*)d_in[12];
  const float* bnv   = (const float*)d_in[13];
  const float* fcw   = (const float*)d_in[14];  // [2, 128, 128]
  const float* fcb   = (const float*)d_in[15];  // [2, 128]
  const float* gfcw  = (const float*)d_in[16];  // [2, 128, 128]
  const float* gfcb  = (const float*)d_in[17];  // [2, 128]
  const float* fw    = (const float*)d_in[18];  // [128, 10]
  const float* fb    = (const float*)d_in[19];  // [10]
  float* out = (float*)d_out;

  const int* erow = eidx;             // edge_index[0]
  const int* ecol = eidx + N_EDGES;   // edge_index[1]

  const long long NH = (long long)N_NODES * HDIM;
  const long long GH = (long long)N_GRAPHS * HDIM;

  // Workspace carve: f32 region (~154MB) then bf16 region (~51MB).
  float* wsf = (float*)d_ws;
  size_t off = 0;
  float* deg = wsf + off; off += N_NODES;      // deg -> deg_inv (in place)
  float* h   = wsf + off; off += NH;           // encoder output (f32)
  float* zA  = wsf + off; off += NH;
  float* zB  = wsf + off; off += NH;
  float* gp  = wsf + off; off += GH;
  float* gq  = wsf + off; off += GH;
  __bf16* bfA = (__bf16*)(wsf + off);          // [N,128] bf16
  __bf16* bfB = bfA + NH;                      // [N,128] bf16
  __bf16* Wp  = bfB + NH;                      // packed weights (<=16K frags)
  (void)ws_size; (void)n_in; (void)in_sizes; (void)out_size;

  // 1) degree + inverse
  launch_zero(deg, N_NODES, stream);
  deg_kernel<<<(N_EDGES + 255) / 256, 256, 0, stream>>>(erow, ew, deg, N_EDGES);
  recip_kernel<<<(N_NODES + 255) / 256, 256, 0, stream>>>(deg, N_NODES);

  // 2) MLP encoder (WMMA): x@w1 relu -> @w2 relu -> @w3 + BN -> h
  dim3 g64 (N_NODES / 16, 64 / 16);
  dim3 g128(N_NODES / 16, HDIM / 16);
  launch_cvt(x, bfA, (long long)N_NODES * C_IN, stream);
  launch_pack(w1, Wp, C_IN, 64, stream);
  wmma_gemm_kernel<1, 1, 128, 64><<<g64, 32, 0, stream>>>(
      bfA, Wp, b1, bfB, nullptr, nullptr, nullptr, nullptr);
  launch_pack(w2, Wp, 64, HDIM, stream);
  wmma_gemm_kernel<1, 1, 64, 128><<<g128, 32, 0, stream>>>(
      bfB, Wp, b2, bfA, nullptr, nullptr, nullptr, nullptr);
  launch_pack(w3, Wp, HDIM, HDIM, stream);
  wmma_gemm_kernel<2, 0, 128, 128><<<g128, 32, 0, stream>>>(
      bfA, Wp, b3, h, bng, bnb, bnm, bnv);

  // 3) DEQ fixed-point: z0 = 0; 10 iterations of scatter + combine
  launch_zero(zA, NH, stream);
  float* cur = zA;
  float* nxt = zB;
  const long long scatterThreads = (long long)N_EDGES * 32;
  for (int it = 0; it < MAX_ITER; ++it) {
    launch_zero(nxt, NH, stream);
    scatter_kernel<<<(unsigned)((scatterThreads + 255) / 256), 256, 0, stream>>>(
        erow, ecol, ew, cur, nxt, N_EDGES);
    combine_kernel<<<(unsigned)((NH + 255) / 256), 256, 0, stream>>>(nxt, deg, h, NH);
    float* t = cur; cur = nxt; nxt = t;
  }

  // 4) node FC stack (WMMA, relu): z -> bf16 -> fc1(bf16 out) -> fc2(f32 out)
  launch_cvt(cur, bfA, NH, stream);
  launch_pack(fcw, Wp, HDIM, HDIM, stream);
  wmma_gemm_kernel<1, 1, 128, 128><<<g128, 32, 0, stream>>>(
      bfA, Wp, fcb, bfB, nullptr, nullptr, nullptr, nullptr);
  launch_pack(fcw + HDIM * HDIM, Wp, HDIM, HDIM, stream);
  wmma_gemm_kernel<1, 0, 128, 128><<<g128, 32, 0, stream>>>(
      bfB, Wp, fcb + HDIM, nxt, nullptr, nullptr, nullptr, nullptr);

  // 5) global_add_pool (nxt holds fc2 output, f32)
  launch_zero(gp, GH, stream);
  const long long poolThreads = (long long)N_NODES * 32;
  pool_kernel<<<(unsigned)((poolThreads + 255) / 256), 256, 0, stream>>>(
      nxt, batch, gp, N_NODES);

  // 6) graph FC stack (WMMA, relu) — 512 rows = 32 row tiles
  dim3 gG(N_GRAPHS / 16, HDIM / 16);
  launch_cvt(gp, bfA, GH, stream);
  launch_pack(gfcw, Wp, HDIM, HDIM, stream);
  wmma_gemm_kernel<1, 1, 128, 128><<<gG, 32, 0, stream>>>(
      bfA, Wp, gfcb, bfB, nullptr, nullptr, nullptr, nullptr);
  launch_pack(gfcw + HDIM * HDIM, Wp, HDIM, HDIM, stream);
  wmma_gemm_kernel<1, 0, 128, 128><<<gG, 32, 0, stream>>>(
      bfB, Wp, gfcb + HDIM, gq, nullptr, nullptr, nullptr, nullptr);

  // 7) final linear + log_softmax
  final_kernel<<<(N_GRAPHS + 255) / 256, 256, 0, stream>>>(gq, fw, fb, out, N_GRAPHS);
}
